// EdgeConvGNN_89412629168423
// MI455X (gfx1250) — matrix operand order repaired and checked by reading, not compiled, not run on hardware
//
#include <hip/hip_runtime.h>
#include <hip/hip_bf16.h>
#include <math.h>

typedef __bf16 bf16_t;
typedef __attribute__((ext_vector_type(16))) __bf16 v16bf;
typedef __attribute__((ext_vector_type(8)))  __bf16 v8bf;
typedef __attribute__((ext_vector_type(8)))  float  v8f;

#define DH 128  // hidden dim (== 2*D_IN)

// ---------------------------------------------------------------- utilities
__global__ void k_fill1(float* __restrict__ p, int n) {
    int i = blockIdx.x * blockDim.x + threadIdx.x;
    if (i < n) p[i] = 1.0f;                 // self-loop contributes 1 to degree
}

__global__ void k_zero(float* __restrict__ p, long n) {
    long i = (long)blockIdx.x * blockDim.x + threadIdx.x;
    if (i < n) p[i] = 0.0f;
}

__global__ void k_count_deg(const int* __restrict__ col, float* __restrict__ deg, int e) {
    int i = blockIdx.x * blockDim.x + threadIdx.x;
    if (i < e) unsafeAtomicAdd(&deg[col[i]], 1.0f);   // global_atomic_add_f32
}

__global__ void k_rsqrt(float* __restrict__ p, int n) {
    int i = blockIdx.x * blockDim.x + threadIdx.x;
    if (i < n) { float d = p[i]; p[i] = (d > 0.0f) ? rsqrtf(d) : 0.0f; }
}

// h[e] = concat(x[src[e]], x[dst[e]])  -> bf16 [EG, 128]
__global__ void k_build_h(const float* __restrict__ x,
                          const int* __restrict__ gsrc, const int* __restrict__ gdst,
                          bf16_t* __restrict__ hbf, long total) {
    long t = (long)blockIdx.x * blockDim.x + threadIdx.x;
    if (t >= total) return;
    int e = (int)(t >> 7), f = (int)(t & 127);
    int node = (f < 64) ? gsrc[e] : gdst[e];
    hbf[t] = (bf16_t)x[(long)node * 64 + (f & 63)];
}

// Wt[n*128 + k] = bf16(W[k*128 + n])   (transpose-pack for contiguous B frags)
__global__ void k_pack_w(const float* __restrict__ W, bf16_t* __restrict__ Wt) {
    int t = blockIdx.x * blockDim.x + threadIdx.x;   // 16384
    int k = t >> 7, n = t & 127;
    Wt[(long)n * DH + k] = (bf16_t)W[t];
}

// -------------------------------------------------- WMMA GEMM  C = A @ W
// A: [M,128] bf16 row-major,  Wt: [128,128] bf16 transposed (Wt[n][k]),  C: [M,128] f32
// block = 256 threads (8 waves); wave w computes 16x16 tile at N = 16*w; K = 4 x 32.
__global__ __launch_bounds__(256) void k_gemm_wmma(
        const bf16_t* __restrict__ A, const bf16_t* __restrict__ Wt,
        float* __restrict__ C, int M) {
    const int wave  = threadIdx.x >> 5;
    const int lane  = threadIdx.x & 31;
    const int lo    = lane & 15;
    const int hiSel = lane >> 4;            // 0 | 1
    const int mBase = blockIdx.x * 16;

    int aRowIdx = mBase + lo; if (aRowIdx >= M) aRowIdx = M - 1;   // clamp tail reads
    const bf16_t* aRow = A  + (long)aRowIdx * DH + hiSel * 8;
    const bf16_t* bCol = Wt + (long)(wave * 16 + lo) * DH + hiSel * 8;

    v8f acc = {};
#pragma unroll
    for (int s = 0; s < 4; ++s) {
        // A frag: lanes 0-15 -> K s*32+{0..7,16..23}; lanes 16-31 -> +8
        v8bf a0 = *(const v8bf*)(aRow + s * 32);
        v8bf a1 = *(const v8bf*)(aRow + s * 32 + 16);
        // B frag: lane holds column n, K s*32+{0..7,16..23}; lanes 16-31 -> +8
        v8bf b0 = *(const v8bf*)(bCol + s * 32);
        v8bf b1 = *(const v8bf*)(bCol + s * 32 + 16);
        v16bf a, b;
#pragma unroll
        for (int i = 0; i < 8; ++i) { a[i] = a0[i]; a[8 + i] = a1[i];
                                      b[i] = b0[i]; b[8 + i] = b1[i]; }
        acc = __builtin_amdgcn_wmma_f32_16x16x32_bf16(
                  false, a, false, b, (short)0, acc, false, false);
    }

    // C layout: VGPR r -> lanes 0-15: M=r, lanes 16-31: M=8+r; N = lane&15
    float* cBase = C + (long)(mBase + hiSel * 8) * DH + wave * 16 + lo;
    if (mBase + 16 <= M) {
        // full tile: block-uniform branch, straight-line stores (no exec churn)
#pragma unroll
        for (int r = 0; r < 8; ++r) cBase[(long)r * DH] = acc[r];
    } else {
#pragma unroll
        for (int r = 0; r < 8; ++r) {
            if (mBase + hiSel * 8 + r < M) cBase[(long)r * DH] = acc[r];
        }
    }
}

// ------------------------------------- scatter: agg[col] += hw[row] * norm
// one thread handles one edge x 32-feature chunk (8 float4 loads, 32 atomics)
__global__ void k_scatter(const int* __restrict__ row, const int* __restrict__ col,
                          const float* __restrict__ dinv,
                          const float* __restrict__ hw, float* __restrict__ agg,
                          long total) {
    long t = (long)blockIdx.x * blockDim.x + threadIdx.x;
    if (t >= total) return;
    int e = (int)(t >> 2), part = (int)(t & 3);
    int r = row[e], c = col[e];
    float nrm = dinv[r] * dinv[c];
    const float4* src = (const float4*)(hw + (long)r * DH + part * 32);
    float* dst = agg + (long)c * DH + part * 32;
#pragma unroll
    for (int j = 0; j < 8; ++j) {
        float4 v = src[j];
        unsafeAtomicAdd(dst + j * 4 + 0, v.x * nrm);
        unsafeAtomicAdd(dst + j * 4 + 1, v.y * nrm);
        unsafeAtomicAdd(dst + j * 4 + 2, v.z * nrm);
        unsafeAtomicAdd(dst + j * 4 + 3, v.w * nrm);
    }
}

// self-loop + bias + relu; write f32 in place, optionally bf16 for next GEMM
__global__ void k_combine(float* __restrict__ agg, const float* __restrict__ hw,
                          const float* __restrict__ dinv, const float* __restrict__ bias,
                          bf16_t* __restrict__ hbf, long total, int writeBf) {
    long t = (long)blockIdx.x * blockDim.x + threadIdx.x;
    if (t >= total) return;
    int e = (int)(t >> 7), f = (int)(t & 127);
    float d = dinv[e];
    float v = agg[t] + hw[t] * d * d + bias[f];
    v = fmaxf(v, 0.0f);
    agg[t] = v;
    if (writeBf) hbf[t] = (bf16_t)v;
}

// sel = h[idx]; out = sigmoid(dot(sel, Wl) + bl)
__global__ void k_final(const float* __restrict__ h, const int* __restrict__ idx01,
                        const float* __restrict__ Wl, const float* __restrict__ bl,
                        float* __restrict__ out) {
    __shared__ float red[DH];
    int f = threadIdx.x;
    int idx = idx01[0];
    red[f] = h[(long)idx * DH + f] * Wl[f];
    __syncthreads();
    for (int s = 64; s > 0; s >>= 1) {
        if (f < s) red[f] += red[f + s];
        __syncthreads();
    }
    if (f == 0) out[0] = 1.0f / (1.0f + expf(-(red[0] + bl[0])));
}

// ---------------------------------------------------------------- launcher
extern "C" void kernel_launch(void* const* d_in, const int* in_sizes, int n_in,
                              void* d_out, int out_size, void* d_ws, size_t ws_size,
                              hipStream_t stream) {
    const float* x     = (const float*)d_in[0];
    const int*   gei   = (const int*)  d_in[1];
    const int*   lgei  = (const int*)  d_in[2];
    const int*   idx01 = (const int*)  d_in[3];
    const float* W0    = (const float*)d_in[4];
    const float* b0    = (const float*)d_in[5];
    const float* W1    = (const float*)d_in[6];
    const float* b1    = (const float*)d_in[7];
    const float* Wl    = (const float*)d_in[8];
    const float* bl    = (const float*)d_in[9];
    float* out = (float*)d_out;

    const int EG  = in_sizes[1] / 2;   // 200000 line-graph nodes
    const int ELG = in_sizes[2] / 2;   // 2000000 line-graph edges
    const long FEG = (long)EG * DH;

    // workspace carve-out (256B aligned)
    char*  base = (char*)d_ws;
    size_t off  = 0;
    auto carve = [&](size_t bytes) { void* p = base + off;
                                     off = (off + bytes + 255) & ~(size_t)255; return p; };
    float*  dinv = (float*) carve((size_t)EG * 4);
    bf16_t* Wt0  = (bf16_t*)carve((size_t)DH * DH * 2);
    bf16_t* Wt1  = (bf16_t*)carve((size_t)DH * DH * 2);
    bf16_t* hbf  = (bf16_t*)carve((size_t)FEG * 2);
    float*  hwf  = (float*) carve((size_t)FEG * 4);
    float*  agg  = (float*) carve((size_t)FEG * 4);

    const int B = 256;
    const int gEG   = (EG + B - 1) / B;
    const int gELG  = (ELG + B - 1) / B;
    const int gFEG  = (int)((FEG + B - 1) / B);
    const int gSCAT = (int)(((long)ELG * 4 + B - 1) / B);
    const int gGEMM = (EG + 15) / 16;

    // degrees -> dinv
    k_fill1    <<<gEG,  B, 0, stream>>>(dinv, EG);
    k_count_deg<<<gELG, B, 0, stream>>>(lgei + ELG, dinv, ELG);
    k_rsqrt    <<<gEG,  B, 0, stream>>>(dinv, EG);

    // features + packed weights
    k_build_h<<<gFEG, B, 0, stream>>>(x, gei, gei + EG, hbf, FEG);
    k_pack_w <<<64,   B, 0, stream>>>(W0, Wt0);
    k_pack_w <<<64,   B, 0, stream>>>(W1, Wt1);

    // ---- layer 0
    k_gemm_wmma<<<gGEMM, B, 0, stream>>>(hbf, Wt0, hwf, EG);
    k_zero     <<<gFEG,  B, 0, stream>>>(agg, FEG);
    k_scatter  <<<gSCAT, B, 0, stream>>>(lgei, lgei + ELG, dinv, hwf, agg, (long)ELG * 4);
    k_combine  <<<gFEG,  B, 0, stream>>>(agg, hwf, dinv, b0, hbf, FEG, 1);

    // ---- layer 1
    k_gemm_wmma<<<gGEMM, B, 0, stream>>>(hbf, Wt1, hwf, EG);
    k_zero     <<<gFEG,  B, 0, stream>>>(agg, FEG);
    k_scatter  <<<gSCAT, B, 0, stream>>>(lgei, lgei + ELG, dinv, hwf, agg, (long)ELG * 4);
    k_combine  <<<gFEG,  B, 0, stream>>>(agg, hwf, dinv, b1, hbf, FEG, 0);

    // ---- head
    k_final<<<1, DH, 0, stream>>>(agg, idx01, Wl, bl, out);
}
